// CNP_5669356834854
// MI455X (gfx1250) — matrix operand order repaired and checked by reading, not compiled, not run on hardware
//
#include <hip/hip_runtime.h>
#include <hip/hip_bf16.h>

// ---------------------------------------------------------------------------
// Problem constants (from the reference):
//   B=2, L=2048, D=1024, H=16, DK=DV=64, EPS=1e-5
// Inputs (d_in order): q,k,v,k_gate,mask,wq,bq,wk,bk,wv,bv,wo,bo,gamma,beta
// Output: fp32 [B, L, D]
// ---------------------------------------------------------------------------

#define CB   2
#define CL   2048
#define CD   1024
#define CH   16
#define CDK  64
#define CM   (CB * CL)      // 4096 rows
#define CHD  (CH * CDK)     // 1024

typedef _Float16 v16h __attribute__((ext_vector_type(16)));
typedef _Float16 v8h  __attribute__((ext_vector_type(8)));
typedef float    v8f  __attribute__((ext_vector_type(8)));
typedef unsigned int u32x4 __attribute__((ext_vector_type(4)));
typedef int          i32x4 __attribute__((ext_vector_type(4)));
typedef int          i32x8 __attribute__((ext_vector_type(8)));

#if defined(__has_builtin)
#if __has_builtin(__builtin_amdgcn_tensor_load_to_lds) && \
    __has_builtin(__builtin_amdgcn_s_wait_tensorcnt)
#define ATTN_HAVE_TDM 1
#endif
#endif
#ifndef ATTN_HAVE_TDM
#define ATTN_HAVE_TDM 0
#endif

static __device__ __forceinline__ v8f vzero8() {
    v8f z;
#pragma unroll
    for (int i = 0; i < 8; ++i) z[i] = 0.0f;
    return z;
}

static __device__ __forceinline__ v16h vconcat(v8h lo, v8h hi) {
    v16h r;
#pragma unroll
    for (int i = 0; i < 8; ++i) { r[i] = lo[i]; r[i + 8] = hi[i]; }
    return r;
}

// ===========================================================================
// GEMM:  out[M,N] = A[M,K] @ W[N,K]^T + bias (+ residual), via WMMA f16.
// Block tile 128x128, K-step 32, 8 waves: wave grid 2(M) x 4(N),
// per-wave 64x32 = 4x2 tiles of 16x16 -> 8 v_wmma per K-step.
// ===========================================================================
template <bool A_IS_F32, bool RESID, bool OUT_F16>
__global__ __launch_bounds__(256) void gemm_bias(
    const void* __restrict__ Av, const float* __restrict__ W,
    const float* __restrict__ bias, const float* __restrict__ resid,
    void* __restrict__ outv, int M, int N, int K)
{
    // stride 40 halves = 80B: multiple of 16B so v8h fragment loads stay aligned
    __shared__ _Float16 As[128][40];
    __shared__ _Float16 Bs[128][40];

    const int tid  = threadIdx.x;
    const int lane = tid & 31;
    const int w    = tid >> 5;
    const int wm   = (w & 1) * 64;   // wave M offset in block tile
    const int wn   = (w >> 1) * 32;  // wave N offset in block tile
    const int m0   = blockIdx.y * 128;
    const int n0   = blockIdx.x * 128;

    v8f acc[4][2];
#pragma unroll
    for (int mi = 0; mi < 4; ++mi)
#pragma unroll
        for (int ni = 0; ni < 2; ++ni) acc[mi][ni] = vzero8();

    const int srow = tid >> 1;        // 0..127: staged row
    const int scg  = (tid & 1) * 16;  // 0 or 16: 16-element chunk

    const int hlo  = (lane < 16) ? 0 : 1;  // lane half

    for (int k0 = 0; k0 < K; k0 += 32) {
        // ---- stage A tile (convert to f16 if needed) ----
        if constexpr (A_IS_F32) {
            const float* src = (const float*)Av + (size_t)(m0 + srow) * K + k0 + scg;
#pragma unroll
            for (int i = 0; i < 4; ++i) {
                float4 f = *(const float4*)(src + i * 4);
                As[srow][scg + i * 4 + 0] = (_Float16)f.x;
                As[srow][scg + i * 4 + 1] = (_Float16)f.y;
                As[srow][scg + i * 4 + 2] = (_Float16)f.z;
                As[srow][scg + i * 4 + 3] = (_Float16)f.w;
            }
        } else {
            const _Float16* src = (const _Float16*)Av + (size_t)(m0 + srow) * K + k0 + scg;
            *(v8h*)&As[srow][scg]     = *(const v8h*)src;
            *(v8h*)&As[srow][scg + 8] = *(const v8h*)(src + 8);
        }
        // ---- stage W tile (rows are N, contiguous K) ----
        {
            const float* src = W + (size_t)(n0 + srow) * K + k0 + scg;
#pragma unroll
            for (int i = 0; i < 4; ++i) {
                float4 f = *(const float4*)(src + i * 4);
                Bs[srow][scg + i * 4 + 0] = (_Float16)f.x;
                Bs[srow][scg + i * 4 + 1] = (_Float16)f.y;
                Bs[srow][scg + i * 4 + 2] = (_Float16)f.z;
                Bs[srow][scg + i * 4 + 3] = (_Float16)f.w;
            }
        }
        __syncthreads();

        // ---- B fragments: col = lane&15, K 0-15 (lanes 0-15) / 16-31 (16-31) ----
        v16h bf[2];
#pragma unroll
        for (int ni = 0; ni < 2; ++ni) {
            const int bn = wn + ni * 16 + (lane & 15);
            const int kb = hlo * 16;
            v8h lo = *(const v8h*)&Bs[bn][kb];
            v8h hi = *(const v8h*)&Bs[bn][kb + 8];
            bf[ni] = vconcat(lo, hi);
        }
        // ---- A fragments + WMMA: row = lane&15, K kb..kb+15 (kb = 16*laneHalf) ----
#pragma unroll
        for (int mi = 0; mi < 4; ++mi) {
            const int ar = wm + mi * 16 + (lane & 15);
            const int kb = hlo * 16;
            v8h lo = *(const v8h*)&As[ar][kb];
            v8h hi = *(const v8h*)&As[ar][kb + 8];
            v16h af = vconcat(lo, hi);
#pragma unroll
            for (int ni = 0; ni < 2; ++ni)
                acc[mi][ni] = __builtin_amdgcn_wmma_f32_16x16x32_f16(
                    false, af, false, bf[ni], (short)0, acc[mi][ni], false, false);
        }
        __syncthreads();
    }

    // ---- epilogue: C layout row = base + r + 8*laneHalf, col = lane&15 ----
#pragma unroll
    for (int mi = 0; mi < 4; ++mi) {
#pragma unroll
        for (int ni = 0; ni < 2; ++ni) {
            const int n  = n0 + wn + ni * 16 + (lane & 15);
            const float bn = bias[n];
#pragma unroll
            for (int r = 0; r < 8; ++r) {
                const int m = m0 + wm + mi * 16 + r + hlo * 8;
                float v = acc[mi][ni][r] + bn;
                if constexpr (RESID) v += resid[(size_t)m * N + n];
                if constexpr (OUT_F16)
                    ((_Float16*)outv)[(size_t)m * N + n] = (_Float16)v;
                else
                    ((float*)outv)[(size_t)m * N + n] = v;
            }
        }
    }
}

// ===========================================================================
// Flash-attention with multiplicative k-gating.
// Grid: (L/128, B*H). 8 waves x 16 q-rows = 128 q-rows per block.
// K-tile = 32 key rows. The K tile is staged LDS-side by the Tensor Data
// Mover (TENSOR_LOAD_TO_LDS, one issue per k-step from wave 0, fenced with
// s_wait_tensorcnt) using D# padding to produce the 72-half LDS row stride.
// S = Q K^T (WMMA), scale * gate * mask, online softmax, O += P V (WMMA).
// gate ([B,H,L,L] fp32, 512MB) is streamed exactly once -> HBM roofline term.
// ===========================================================================
__global__ __launch_bounds__(256) void attn_kernel(
    const _Float16* __restrict__ Qh, const _Float16* __restrict__ Kh,
    const _Float16* __restrict__ Vh, const float* __restrict__ gate,
    const unsigned char* __restrict__ mask, _Float16* __restrict__ AO)
{
    __shared__ _Float16 Ks[32][72];      // [krow][dk]   144B row stride (16B mult)
    __shared__ _Float16 Vst[64][48];     // [dv][krow]   96B row stride  (16B mult)
    __shared__ _Float16 Ps[8][16][40];   // per-wave P:  80B row stride  (16B mult)

    const int tid  = threadIdx.x;
    const int lane = tid & 31;
    const int w    = tid >> 5;
    const int hlo  = (lane < 16) ? 0 : 1;
    const int bh   = blockIdx.y;
    const int b    = bh >> 4;      // / H
    const int h    = bh & 15;      // % H
    const int q0   = blockIdx.x * 128;
    const int qb   = q0 + w * 16;  // wave's first q row

    // ---- Q fragments held in registers for the whole k loop ----
    v16h qf[2];
    {
        const int qrow = qb + (lane & 15);
        const _Float16* qp = Qh + ((size_t)(b * CL + qrow)) * CHD + h * CDK;
#pragma unroll
        for (int c = 0; c < 2; ++c) {
            const int ko = c * 32 + hlo * 16;
            v8h lo = *(const v8h*)(qp + ko);
            v8h hi = *(const v8h*)(qp + ko + 8);
            qf[c] = vconcat(lo, hi);
        }
    }

    v8f oacc[4];
#pragma unroll
    for (int vi = 0; vi < 4; ++vi) oacc[vi] = vzero8();
    float mrow[8], lrow[8];
#pragma unroll
    for (int r = 0; r < 8; ++r) { mrow[r] = -1e30f; lrow[r] = 0.0f; }

    const int tkr = tid >> 3;        // 0..31 key row within tile
    const int tdg = (tid & 7) * 8;   // 8-element chunk of d

    for (int kt = 0; kt < CL; kt += 32) {
        // ---- stage K tile [32][64] via the Tensor Data Mover ----
#if ATTN_HAVE_TDM
        if (w == 0) {
            // D# per CDNA5 ISA:
            //   2D tile 64(x) x 32(y) of 2-byte elements, row stride 1024 elems,
            //   LDS padding: +4 DWORDs (8 halves) after every 32 DWORDs (64 halves)
            //   -> LDS row stride 72 halves, matching Ks[][72].
            const uint64_t ga = (uint64_t)(uintptr_t)
                (Kh + ((size_t)(b * CL + kt)) * CHD + h * CDK);
            u32x4 g0;
            g0[0] = 1u;                                  // count=1, user descriptor
            g0[1] = (uint32_t)(uintptr_t)(&Ks[0][0]);    // lds_addr (bytes)
            g0[2] = (uint32_t)ga;                        // global_addr[31:0]
            g0[3] = (uint32_t)((ga >> 32) & 0x1FFFFFFull) | (2u << 30); // [56:32], type=2
            i32x8 g1;
            g1[0] = (1 << 16)            // data_size = 2 bytes
                  | (1 << 20)            // pad_enable
                  | (4 << 22)            // pad_interval: 32 DWORDs (one 64-half row)
                  | (3 << 25);           // pad_amount: 4 DWORDs (8 halves)
            g1[1] = (int)(64u << 16);    // tensor_dim0 = 64 (bits 79:48)
            g1[2] = (int)(32u << 16);    // tensor_dim1 = 32 (bits 111:80)
            g1[3] = (int)(64u << 16);    // tile_dim0 = 64 (bits 127:112)
            g1[4] = 32;                  // tile_dim1 = 32, tile_dim2 = 0
            g1[5] = 1024;                // tensor_dim0_stride = 1024 elements
            g1[6] = 0;
            g1[7] = 0;
            i32x4 gz; gz[0] = 0; gz[1] = 0; gz[2] = 0; gz[3] = 0;
#if __clang_major__ >= 23
            i32x8 gz8;
#pragma unroll
            for (int i = 0; i < 8; ++i) gz8[i] = 0;
            __builtin_amdgcn_tensor_load_to_lds(g0, g1, gz, gz, gz8, 0);
#else
            __builtin_amdgcn_tensor_load_to_lds(g0, g1, gz, gz, 0);
#endif
            __builtin_amdgcn_s_wait_tensorcnt(0);
        }
#else
        {
            const size_t koff = ((size_t)(b * CL + kt + tkr)) * CHD + h * CDK + tdg;
            *(v8h*)&Ks[tkr][tdg] = *(const v8h*)(Kh + koff);
        }
#endif
        // ---- stage V tile transposed [64][32] (TDM cannot transpose) ----
        {
            const size_t rowoff = ((size_t)(b * CL + kt + tkr)) * CHD + h * CDK + tdg;
            v8h vv = *(const v8h*)(Vh + rowoff);
#pragma unroll
            for (int j = 0; j < 8; ++j) Vst[tdg + j][tkr] = vv[j];
        }
        __syncthreads();

        // prefetch next gate tile into cache (global_prefetch_b8)
        if (kt + 32 < CL) {
            const float* gp = gate + ((size_t)bh * CL + (qb + (lane & 15))) * CL
                                   + (kt + 32) + hlo * 16;
            __builtin_prefetch(gp, 0, 1);
        }

        // ---- S = Q K^T : two 16x16 N-subtiles, contraction 64 in 2 chunks ----
        v8f s0 = vzero8(), s1 = vzero8();
#pragma unroll
        for (int c = 0; c < 2; ++c) {
            const int kb = c * 32 + hlo * 16;
            {
                const int bn = (lane & 15);
                v8h lo = *(const v8h*)&Ks[bn][kb];
                v8h hi = *(const v8h*)&Ks[bn][kb + 8];
                s0 = __builtin_amdgcn_wmma_f32_16x16x32_f16(
                    false, qf[c], false, vconcat(lo, hi), (short)0, s0, false, false);
            }
            {
                const int bn = 16 + (lane & 15);
                v8h lo = *(const v8h*)&Ks[bn][kb];
                v8h hi = *(const v8h*)&Ks[bn][kb + 8];
                s1 = __builtin_amdgcn_wmma_f32_16x16x32_f16(
                    false, qf[c], false, vconcat(lo, hi), (short)0, s1, false, false);
            }
        }

        // ---- scale (1/sqrt(64)), multiplicative gate, additive mask ----
        float sv[2][8];
#pragma unroll
        for (int r = 0; r < 8; ++r) {
            const int qrow   = qb + r + hlo * 8;
            const size_t gro = ((size_t)bh * CL + qrow) * CL;
            const size_t mro = ((size_t)b * CL + qrow) * CL;
#pragma unroll
            for (int ni = 0; ni < 2; ++ni) {
                const int col = kt + ni * 16 + (lane & 15);
                const float g = gate[gro + col];
                float val = (ni == 0 ? s0[r] : s1[r]) * 0.125f * g;
                if (mask[mro + col]) val = -1e30f;
                sv[ni][r] = val;
            }
        }

        // ---- online softmax (row reductions across each half-wave of 16) ----
#pragma unroll
        for (int r = 0; r < 8; ++r) {
            float rm = fmaxf(sv[0][r], sv[1][r]);
#pragma unroll
            for (int off = 8; off >= 1; off >>= 1)
                rm = fmaxf(rm, __shfl_xor(rm, off, 16));
            const float mn   = fmaxf(mrow[r], rm);
            const float corr = __expf(mrow[r] - mn);
            const float p0   = __expf(sv[0][r] - mn);
            const float p1   = __expf(sv[1][r] - mn);
            float rs = p0 + p1;
#pragma unroll
            for (int off = 8; off >= 1; off >>= 1)
                rs += __shfl_xor(rs, off, 16);
            lrow[r] = lrow[r] * corr + rs;
            mrow[r] = mn;
#pragma unroll
            for (int vi = 0; vi < 4; ++vi) oacc[vi][r] *= corr;
            // spill P (C layout) to per-wave LDS for A-fragment reshape
            const int prow = r + hlo * 8;
            Ps[w][prow][(lane & 15)]      = (_Float16)p0;
            Ps[w][prow][16 + (lane & 15)] = (_Float16)p1;
        }
        __asm__ volatile("" ::: "memory");   // keep ds store->load order
        __builtin_amdgcn_wave_barrier();     // per-wave region: DS in-order suffices

        // ---- P A-fragment (16x32) ----
        v16h pf;
        {
            const int pr = lane & 15;
            const int kb = hlo * 16;
            v8h lo = *(const v8h*)&Ps[w][pr][kb];
            v8h hi = *(const v8h*)&Ps[w][pr][kb + 8];
            pf = vconcat(lo, hi);
        }
        // ---- O += P V : 4 dv-subtiles, contraction = 32 key rows ----
#pragma unroll
        for (int vi = 0; vi < 4; ++vi) {
            const int dv = vi * 16 + (lane & 15);
            const int kb = hlo * 16;
            v8h lo = *(const v8h*)&Vst[dv][kb];
            v8h hi = *(const v8h*)&Vst[dv][kb + 8];
            oacc[vi] = __builtin_amdgcn_wmma_f32_16x16x32_f16(
                false, pf, false, vconcat(lo, hi), (short)0, oacc[vi], false, false);
        }
        __syncthreads();
    }

    // ---- normalize and store attention output (f16, [B,L,H,DV]) ----
#pragma unroll
    for (int r = 0; r < 8; ++r) {
        const float inv = 1.0f / lrow[r];
        const int qrow  = qb + r + hlo * 8;
        _Float16* ob = AO + ((size_t)(b * CL + qrow)) * CHD + h * CDK;
#pragma unroll
        for (int vi = 0; vi < 4; ++vi)
            ob[vi * 16 + (lane & 15)] = (_Float16)(oacc[vi][r] * inv);
    }
}

// ===========================================================================
// Row LayerNorm over D=1024, in place on d_out. One block per row.
// ===========================================================================
__global__ __launch_bounds__(256) void layernorm_kernel(
    float* __restrict__ io, const float* __restrict__ gamma,
    const float* __restrict__ beta)
{
    __shared__ float red[16];
    const int row = blockIdx.x;
    const int tid = threadIdx.x;
    float* p = io + (size_t)row * CD + tid * 4;
    float4 x = *(const float4*)p;
    float s = x.x + x.y + x.z + x.w;
    float q = x.x * x.x + x.y * x.y + x.z * x.z + x.w * x.w;
#pragma unroll
    for (int off = 16; off >= 1; off >>= 1) {
        s += __shfl_xor(s, off, 32);
        q += __shfl_xor(q, off, 32);
    }
    const int w = tid >> 5, lane = tid & 31;
    if (lane == 0) { red[w] = s; red[8 + w] = q; }
    __syncthreads();
    float ts = 0.0f, tq = 0.0f;
#pragma unroll
    for (int i = 0; i < 8; ++i) { ts += red[i]; tq += red[8 + i]; }
    const float mu   = ts * (1.0f / CD);
    const float var  = tq * (1.0f / CD) - mu * mu;
    const float rstd = rsqrtf(var + 1e-5f);
    float4 g  = *(const float4*)(gamma + tid * 4);
    float4 be = *(const float4*)(beta + tid * 4);
    float4 y;
    y.x = (x.x - mu) * rstd * g.x + be.x;
    y.y = (x.y - mu) * rstd * g.y + be.y;
    y.z = (x.z - mu) * rstd * g.z + be.z;
    y.w = (x.w - mu) * rstd * g.w + be.w;
    *(float4*)p = y;
}

// ===========================================================================
extern "C" void kernel_launch(void* const* d_in, const int* in_sizes, int n_in,
                              void* d_out, int out_size, void* d_ws, size_t ws_size,
                              hipStream_t stream)
{
    const float* q     = (const float*)d_in[0];
    const float* k     = (const float*)d_in[1];
    const float* v     = (const float*)d_in[2];
    const float* gate  = (const float*)d_in[3];
    const unsigned char* mask = (const unsigned char*)d_in[4];
    const float* wq = (const float*)d_in[5];
    const float* bq = (const float*)d_in[6];
    const float* wk = (const float*)d_in[7];
    const float* bk = (const float*)d_in[8];
    const float* wv = (const float*)d_in[9];
    const float* bv = (const float*)d_in[10];
    const float* wo = (const float*)d_in[11];
    const float* bo = (const float*)d_in[12];
    const float* gm = (const float*)d_in[13];
    const float* bt = (const float*)d_in[14];

    // workspace: Qh | Kh | Vh | AO, each CM*CHD f16 (8 MB) -> 32 MB total
    const size_t MH = (size_t)CM * CHD;
    _Float16* Qh = (_Float16*)d_ws;
    _Float16* Kh = Qh + MH;
    _Float16* Vh = Kh + MH;
    _Float16* AO = Vh + MH;

    dim3 gg(CD / 128, CM / 128);  // (8, 32)
    gemm_bias<true, false, true><<<gg, 256, 0, stream>>>(q, wq, bq, nullptr, Qh, CM, CHD, CD);
    gemm_bias<true, false, true><<<gg, 256, 0, stream>>>(k, wk, bk, nullptr, Kh, CM, CHD, CD);
    gemm_bias<true, false, true><<<gg, 256, 0, stream>>>(v, wv, bv, nullptr, Vh, CM, CHD, CD);

    attn_kernel<<<dim3(CL / 128, CB * CH), 256, 0, stream>>>(Qh, Kh, Vh, gate, mask, AO);

    gemm_bias<false, true, false><<<gg, 256, 0, stream>>>(AO, wo, bo, q, (float*)d_out,
                                                          CM, CD, CHD);
    layernorm_kernel<<<CM, 256, 0, stream>>>((float*)d_out, gm, bt);
}
